// MultiHeadSelfAttention_13391708029968
// MI455X (gfx1250) — compile-verified, matrix operand order checked by reading, not run on hardware
//
#include <hip/hip_runtime.h>

// ---------------- WMMA types & helpers (gfx1250, wave32) ----------------
typedef __attribute__((ext_vector_type(16))) __bf16 v16bf;
typedef __attribute__((ext_vector_type(8)))  float  v8f;

union FragU { v16bf v; unsigned u[8]; uint4 q[2]; };

__device__ __forceinline__ unsigned bf16_rne(float a) {
  unsigned u = __float_as_uint(a);
  return (u + 0x7FFFu + ((u >> 16) & 1u)) >> 16;
}
__device__ __forceinline__ unsigned bf16x2_from_f32(float a, float b) {
  return (bf16_rne(a) & 0xFFFFu) | (bf16_rne(b) << 16);
}

// Fragment of v_wmma_f32_16x16x32_bf16 for row-major bf16 source (A, or B as [N][K]):
// lane (m = lane&15, half = lane>>4): VGPRs 0..3 = K 8h..8h+7, VGPRs 4..7 = K 16+8h..16+8h+7
// => two contiguous 16-byte runs per lane.
__device__ __forceinline__ v16bf load_frag_bf16(const unsigned short* __restrict__ src,
                                                int ld, int lane) {
  const int m = lane & 15, half = lane >> 4;
  const unsigned short* p = src + (long)m * ld + 8 * half;
  FragU f;
  f.q[0] = *reinterpret_cast<const uint4*>(p);
  f.q[1] = *reinterpret_cast<const uint4*>(p + 16);
  return f.v;
}

__device__ __forceinline__ v8f wmma_bf16(v16bf a, v16bf b, v8f c) {
  // (neg_a, A, neg_b, B, c_mod, C, reuse_a, reuse_b)
  return __builtin_amdgcn_wmma_f32_16x16x32_bf16(false, a, false, b, (short)0, c, false, false);
}

// ---------------- Problem constants ----------------
#define BATCH 4
#define SEQ   2048
#define DMODEL 512
#define NHEAD 4
#define HDIM  128
#define ROWS  (BATCH * SEQ)   // 8192

// ---------------- Kernel 0: fp32 -> bf16 bulk convert (pairs) ----------------
__global__ void __launch_bounds__(256) cvt_bf16_kernel(const float* __restrict__ in,
                                                       unsigned* __restrict__ out,
                                                       int npairs) {
  int i = blockIdx.x * 256 + threadIdx.x;
  if (i < npairs) {
    float2 f = reinterpret_cast<const float2*>(in)[i];
    out[i] = bf16x2_from_f32(f.x, f.y);
  }
}

// ---------------- Kernel 1: QKV projection  Y = x @ W^T + b  (bf16 out) ------
// Wave computes a 16x64 tile; fragments are double-buffered in registers so the
// loads for step k+32 overlap the 4 WMMAs of step k.
// vtranspose=0: out layout [B][H][L][Hd] ; vtranspose=1: out layout [B][H][Hd][L]
__global__ void __launch_bounds__(128) qkv_proj_kernel(
    const unsigned short* __restrict__ xb, const unsigned short* __restrict__ Wb,
    const float* __restrict__ bias, unsigned short* __restrict__ out,
    float scale, int vtranspose)
{
  const int lane = threadIdx.x;                 // 0..31
  const int rt = blockIdx.x;                    // 16-row tile of 8192
  const int wt = blockIdx.y * 4 + threadIdx.y;  // 64-col group [0,8)
  const unsigned short* arow = xb + (long)rt * 16 * DMODEL;
  const unsigned short* brow = Wb + (long)wt * 64 * DMODEL;  // W is [out][in]
  v8f acc[4];
#pragma unroll
  for (int t = 0; t < 4; ++t) acc[t] = (v8f){};

  v16bf a_cur = load_frag_bf16(arow, DMODEL, lane);
  v16bf b_cur[4];
#pragma unroll
  for (int t = 0; t < 4; ++t) b_cur[t] = load_frag_bf16(brow + (long)t * 16 * DMODEL, DMODEL, lane);

#pragma unroll 4
  for (int k = 0; k < DMODEL; k += 32) {
    const int kn = (k + 32) & (DMODEL - 1);     // wraps on last iter (harmless prefetch)
    v16bf a_nxt = load_frag_bf16(arow + kn, DMODEL, lane);
    v16bf b_nxt[4];
#pragma unroll
    for (int t = 0; t < 4; ++t) b_nxt[t] = load_frag_bf16(brow + (long)t * 16 * DMODEL + kn, DMODEL, lane);
#pragma unroll
    for (int t = 0; t < 4; ++t) acc[t] = wmma_bf16(a_cur, b_cur[t], acc[t]);
    a_cur = a_nxt;
#pragma unroll
    for (int t = 0; t < 4; ++t) b_cur[t] = b_nxt[t];
  }

  const int half = lane >> 4;
  const int col = lane & 15;
#pragma unroll
  for (int t = 0; t < 4; ++t) {
    const int n = wt * 64 + t * 16 + col;       // output column in [0,512)
    const int h = n >> 7, d = n & (HDIM - 1);
    const float bn = bias[n];
#pragma unroll
    for (int r = 0; r < 8; ++r) {
      int mg = rt * 16 + r + 8 * half;          // global row in [0,8192)
      int bi = mg >> 11, l = mg & (SEQ - 1);
      float val = (acc[t][r] + bn) * scale;
      long idx = vtranspose
               ? (((long)(bi * NHEAD + h) * HDIM + d) * SEQ + l)
               : (((long)(bi * NHEAD + h) * SEQ + l) * HDIM + d);
      out[idx] = (unsigned short)bf16_rne(val);
    }
  }
}

// ---------------- Kernel 2: flash attention ----------------
// Q,K: bf16 [B][H][L][Hd]; VT: bf16 [B][H][Hd][L]; ctx out: bf16 [B*L][512]
__global__ void __launch_bounds__(128) attn_kernel(
    const unsigned short* __restrict__ Q, const unsigned short* __restrict__ K,
    const unsigned short* __restrict__ VT, unsigned short* __restrict__ ctx)
{
  __shared__ __align__(16) unsigned short plds[4][16 * 32]; // P tile per wave
  const int lane = threadIdx.x;
  const int w = threadIdx.y;
  const int qt = blockIdx.x * 4 + w;            // q tile [0,128)
  const int h = blockIdx.y, b = blockIdx.z;
  const int bh = b * NHEAD + h;
  const unsigned short* Qt = Q  + ((long)bh * SEQ + qt * 16) * HDIM;
  const unsigned short* Kb = K  + (long)bh * SEQ * HDIM;
  const unsigned short* Vb = VT + (long)bh * HDIM * SEQ;

  v16bf aq[4];
#pragma unroll
  for (int kk = 0; kk < 4; ++kk) aq[kk] = load_frag_bf16(Qt + kk * 32, HDIM, lane);

  v8f O[8];
#pragma unroll
  for (int c = 0; c < 8; ++c) O[c] = (v8f){};
  float mrow[8], lrow[8];
#pragma unroll
  for (int r = 0; r < 8; ++r) { mrow[r] = -1e30f; lrow[r] = 0.f; }

  unsigned short* P = &plds[w][0];
  const int half = lane >> 4;
  const int col = lane & 15;

  for (int jb = 0; jb < SEQ; jb += 32) {
    // prefetch next K block (speculative, clamped index)
    {
      int jn = (jb + 32) & (SEQ - 1);
      __builtin_prefetch(Kb + (long)jn * HDIM, 0, 0);
    }
    // scores: two 16x16 tiles over 32 key columns
    v8f s0 = {}, s1 = {};
#pragma unroll
    for (int kk = 0; kk < 4; ++kk) {
      v16bf b0 = load_frag_bf16(Kb + (long)jb * HDIM + kk * 32, HDIM, lane);
      s0 = wmma_bf16(aq[kk], b0, s0);
    }
#pragma unroll
    for (int kk = 0; kk < 4; ++kk) {
      v16bf b1 = load_frag_bf16(Kb + (long)(jb + 16) * HDIM + kk * 32, HDIM, lane);
      s1 = wmma_bf16(aq[kk], b1, s1);
    }
    // issue V-fragment loads now; the softmax VALU work below covers their latency
    v16bf bv[8];
#pragma unroll
    for (int c = 0; c < 8; ++c)
      bv[c] = load_frag_bf16(Vb + (long)(c * 16) * SEQ + jb, SEQ, lane);

    // online softmax per row (row m = r + 8*half lives across 16 lanes of a half-wave)
#pragma unroll
    for (int r = 0; r < 8; ++r) {
      float v = fmaxf(s0[r], s1[r]);
      v = fmaxf(v, __shfl_xor(v, 1, 32));
      v = fmaxf(v, __shfl_xor(v, 2, 32));
      v = fmaxf(v, __shfl_xor(v, 4, 32));
      v = fmaxf(v, __shfl_xor(v, 8, 32));
      float mnew  = fmaxf(mrow[r], v);
      float alpha = __expf(mrow[r] - mnew);
      float p0 = __expf(s0[r] - mnew);
      float p1 = __expf(s1[r] - mnew);
      float rs = p0 + p1;
      rs += __shfl_xor(rs, 1, 32);
      rs += __shfl_xor(rs, 2, 32);
      rs += __shfl_xor(rs, 4, 32);
      rs += __shfl_xor(rs, 8, 32);
      lrow[r] = lrow[r] * alpha + rs;
      mrow[r] = mnew;
#pragma unroll
      for (int c = 0; c < 8; ++c) O[c][r] *= alpha;
      int m = r + 8 * half;
      P[m * 32 + col]      = (unsigned short)bf16_rne(p0);
      P[m * 32 + 16 + col] = (unsigned short)bf16_rne(p1);
    }
    // P (16x32) x V (32x128): A from LDS-restriped P, B from pre-loaded V frags
    v16bf ap = load_frag_bf16(P, 32, lane);
#pragma unroll
    for (int c = 0; c < 8; ++c) O[c] = wmma_bf16(ap, bv[c], O[c]);
  }

  // normalize, store ctx as bf16 [B*L][512]
#pragma unroll
  for (int r = 0; r < 8; ++r) {
    float inv = 1.0f / lrow[r];
    int q = qt * 16 + r + 8 * half;
#pragma unroll
    for (int c = 0; c < 8; ++c) {
      float val = O[c][r] * inv;
      int n = h * HDIM + c * 16 + col;
      ctx[((long)b * SEQ + q) * DMODEL + n] = (unsigned short)bf16_rne(val);
    }
  }
}

// ---------------- Kernel 3: output projection  out = ctx @ Wo^T + bo --------
// Same double-buffered 16x64-per-wave scheme; fp32 output + bias.
__global__ void __launch_bounds__(128) out_proj_kernel(
    const unsigned short* __restrict__ ctx, const unsigned short* __restrict__ Wob,
    const float* __restrict__ bo, float* __restrict__ out)
{
  const int lane = threadIdx.x;
  const int rt = blockIdx.x;
  const int wt = blockIdx.y * 4 + threadIdx.y;
  const unsigned short* arow = ctx + (long)rt * 16 * DMODEL;
  const unsigned short* brow = Wob + (long)wt * 64 * DMODEL;
  v8f acc[4];
#pragma unroll
  for (int t = 0; t < 4; ++t) acc[t] = (v8f){};

  v16bf a_cur = load_frag_bf16(arow, DMODEL, lane);
  v16bf b_cur[4];
#pragma unroll
  for (int t = 0; t < 4; ++t) b_cur[t] = load_frag_bf16(brow + (long)t * 16 * DMODEL, DMODEL, lane);

#pragma unroll 4
  for (int k = 0; k < DMODEL; k += 32) {
    const int kn = (k + 32) & (DMODEL - 1);
    v16bf a_nxt = load_frag_bf16(arow + kn, DMODEL, lane);
    v16bf b_nxt[4];
#pragma unroll
    for (int t = 0; t < 4; ++t) b_nxt[t] = load_frag_bf16(brow + (long)t * 16 * DMODEL + kn, DMODEL, lane);
#pragma unroll
    for (int t = 0; t < 4; ++t) acc[t] = wmma_bf16(a_cur, b_cur[t], acc[t]);
    a_cur = a_nxt;
#pragma unroll
    for (int t = 0; t < 4; ++t) b_cur[t] = b_nxt[t];
  }

  const int half = lane >> 4;
  const int col = lane & 15;
#pragma unroll
  for (int t = 0; t < 4; ++t) {
    const int n = wt * 64 + t * 16 + col;
    const float bn = bo[n];
#pragma unroll
    for (int r = 0; r < 8; ++r) {
      int mg = rt * 16 + r + 8 * half;
      out[(long)mg * DMODEL + n] = acc[t][r] + bn;
    }
  }
}

// ---------------- Host launcher ----------------
extern "C" void kernel_launch(void* const* d_in, const int* in_sizes, int n_in,
                              void* d_out, int out_size, void* d_ws, size_t ws_size,
                              hipStream_t stream) {
  (void)in_sizes; (void)n_in; (void)out_size; (void)ws_size;
  const float* x  = (const float*)d_in[0];
  const float* Wq = (const float*)d_in[1];
  const float* bq = (const float*)d_in[2];
  const float* Wk = (const float*)d_in[3];
  const float* bk = (const float*)d_in[4];
  const float* Wv = (const float*)d_in[5];
  const float* bv = (const float*)d_in[6];
  const float* Wo = (const float*)d_in[7];
  const float* bo = (const float*)d_in[8];
  float* out = (float*)d_out;

  const size_t xelems = (size_t)ROWS * DMODEL;     // 4,194,304
  const size_t welems = (size_t)DMODEL * DMODEL;   //   262,144
  unsigned short* p   = (unsigned short*)d_ws;
  unsigned short* Qb  = p;               p += xelems;
  unsigned short* Kb  = p;               p += xelems;
  unsigned short* Vtb = p;               p += xelems;
  unsigned short* Ctx = p;               p += xelems;
  unsigned short* xbb = p;               p += xelems;
  unsigned short* Wqb = p;               p += welems;
  unsigned short* Wkb = p;               p += welems;
  unsigned short* Wvb = p;               p += welems;
  unsigned short* Wob = p;               p += welems;   // total ~44 MB

  // fp32 -> bf16 conversions (pairs)
  {
    int nx = (int)(xelems / 2), nw = (int)(welems / 2);
    hipLaunchKernelGGL(cvt_bf16_kernel, dim3((nx + 255) / 256), dim3(256), 0, stream, x,  (unsigned*)xbb, nx);
    hipLaunchKernelGGL(cvt_bf16_kernel, dim3((nw + 255) / 256), dim3(256), 0, stream, Wq, (unsigned*)Wqb, nw);
    hipLaunchKernelGGL(cvt_bf16_kernel, dim3((nw + 255) / 256), dim3(256), 0, stream, Wk, (unsigned*)Wkb, nw);
    hipLaunchKernelGGL(cvt_bf16_kernel, dim3((nw + 255) / 256), dim3(256), 0, stream, Wv, (unsigned*)Wvb, nw);
    hipLaunchKernelGGL(cvt_bf16_kernel, dim3((nw + 255) / 256), dim3(256), 0, stream, Wo, (unsigned*)Wob, nw);
  }

  dim3 blk(32, 4);                                // 4 wave32 per block
  dim3 gproj(ROWS / 16, 2);                       // 512 row tiles x 8 col groups (4/block)
  const float qscale = 0.08838834764831845f;      // 1/sqrt(HDIM)

  hipLaunchKernelGGL(qkv_proj_kernel, gproj, blk, 0, stream, xbb, Wqb, bq, Qb,  qscale, 0);
  hipLaunchKernelGGL(qkv_proj_kernel, gproj, blk, 0, stream, xbb, Wkb, bk, Kb,  1.0f,   0);
  hipLaunchKernelGGL(qkv_proj_kernel, gproj, blk, 0, stream, xbb, Wvb, bv, Vtb, 1.0f,   1);

  dim3 gattn(SEQ / 16 / 4, NHEAD, BATCH);         // (32, 4, 4)
  hipLaunchKernelGGL(attn_kernel, gattn, blk, 0, stream, Qb, Kb, Vtb, Ctx);

  hipLaunchKernelGGL(out_proj_kernel, gproj, blk, 0, stream, Ctx, Wob, bo, out);
}
